// MCN_8005819040186
// MI455X (gfx1250) — matrix operand order repaired and checked by reading, not compiled, not run on hardware
//
#include <hip/hip_runtime.h>
#include <math.h>

typedef __attribute__((ext_vector_type(16))) _Float16 v16h;
typedef __attribute__((ext_vector_type(8)))  float    v8f;
typedef __attribute__((ext_vector_type(4)))  float    f32x4;

#define BATCH   64
#define NITEM   7
#define NPAIR   28
#define RELCOLS 112

__device__ __forceinline__ float wave_sum(float v) {
#pragma unroll
  for (int off = 16; off > 0; off >>= 1) v += __shfl_xor(v, off, 32);
  return v;
}

// ---------------------------------------------------------------------------
// 1) Spatial mean-pool: one wave per (b, item, channel) plane.
//    Bandwidth-critical: float4 non-temporal streaming loads.
// ---------------------------------------------------------------------------
__global__ void pool_kernel(const float* __restrict__ in, float* __restrict__ out,
                            int nplanes, int len4, float inv_n) {
  int wave = (blockIdx.x * blockDim.x + threadIdx.x) >> 5;
  int lane = threadIdx.x & 31;
  if (wave >= nplanes) return;                       // uniform per wave
  const f32x4* p = reinterpret_cast<const f32x4*>(in) + (size_t)wave * len4;
  float s = 0.f;
  for (int i = lane; i < len4; i += 32) {
    f32x4 v = __builtin_nontemporal_load(p + i);
    s += v.x + v.y + v.z + v.w;
  }
  s = wave_sum(s);
  if (lane == 0) out[wave] = s * inv_n;
}

// ---------------------------------------------------------------------------
// 2) Masked pair cosine: one wave per (b, pair).
// ---------------------------------------------------------------------------
__global__ void paircos_kernel(const float* __restrict__ rep,
                               const float* __restrict__ mask,
                               float* __restrict__ rel,
                               int E, int relu_mask, int col_off) {
  int gw   = (blockIdx.x * blockDim.x + threadIdx.x) >> 5;  // 0..B*28-1
  int lane = threadIdx.x & 31;
  int b = gw / NPAIR, p = gw % NPAIR;
  int pi = 0, pj = 0, cnt = 0;
  for (int a = 0; a < NITEM; ++a)
    for (int c = a; c < NITEM; ++c) { if (cnt == p) { pi = a; pj = c; } ++cnt; }
  const float* xi = rep + ((size_t)b * NITEM + pi) * E;
  const float* xj = rep + ((size_t)b * NITEM + pj) * E;
  const float* m  = mask + (size_t)p * E;
  float sxx = 0.f, syy = 0.f, sxy = 0.f;
  for (int e = lane; e < E; e += 32) {
    float mv = m[e];
    if (relu_mask) mv = fmaxf(mv, 0.f);
    float a = xi[e] * mv, c = xj[e] * mv;
    sxx += a * a; syy += c * c; sxy += a * c;
  }
  sxx = wave_sum(sxx); syy = wave_sum(syy); sxy = wave_sum(sxy);
  if (lane == 0) {
    float ni = fmaxf(sqrtf(sxx), 1e-12f);
    float nj = fmaxf(sqrtf(syy), 1e-12f);
    rel[(size_t)b * RELCOLS + col_off + p] = sxy / (ni * nj);
  }
}

// ---------------------------------------------------------------------------
// 3) BatchNorm (training stats, biased var) + fp16 staging of A and W1,
//    K padded 112 -> 128 with zeros. One block of 256 threads.
// ---------------------------------------------------------------------------
__global__ void bn_stage_kernel(const float* __restrict__ rel,
                                const float* __restrict__ gamma,
                                const float* __restrict__ beta,
                                const float* __restrict__ W1,
                                _Float16* __restrict__ Astage,   // [64][128]
                                _Float16* __restrict__ W1h) {    // [128][128]
  int t = threadIdx.x;
  if (t < 128) {
    if (t < RELCOLS) {
      float mu = 0.f;
      for (int r = 0; r < BATCH; ++r) mu += rel[r * RELCOLS + t];
      mu *= (1.0f / BATCH);
      float var = 0.f;
      for (int r = 0; r < BATCH; ++r) {
        float d = rel[r * RELCOLS + t] - mu; var += d * d;
      }
      var *= (1.0f / BATCH);
      float inv = 1.0f / sqrtf(var + 1e-5f);
      float g = gamma[t], bb = beta[t];
      for (int r = 0; r < BATCH; ++r)
        Astage[r * 128 + t] = (_Float16)((rel[r * RELCOLS + t] - mu) * inv * g + bb);
    } else {
      for (int r = 0; r < BATCH; ++r) Astage[r * 128 + t] = (_Float16)0.f;
    }
  }
  for (int idx = t; idx < 128 * 128; idx += 256) {
    int k = idx >> 7, n = idx & 127;
    W1h[idx] = (_Float16)((k < RELCOLS) ? W1[k * 128 + n] : 0.f);
  }
}

// ---------------------------------------------------------------------------
// 4) Layer 1 via V_WMMA_F32_16X16X32_F16: 32 waves, one 16x16 tile each.
//    M=64 (4 tiles) x N=128 (8 tiles), K loop = 4 x 32.
// ---------------------------------------------------------------------------
__global__ void mlp1_wmma_kernel(const _Float16* __restrict__ A,   // [64][128]
                                 const _Float16* __restrict__ Bm,  // [128][128] K-major
                                 const float* __restrict__ b1,
                                 float* __restrict__ H) {          // [64][128]
  int lane = threadIdx.x & 31;
  int gw   = blockIdx.x * 8 + (threadIdx.x >> 5);  // 0..31
  int mt = gw >> 3, nt = gw & 7;
  int m  = lane & 15;
  int hi = lane >> 4;                              // lane-group select
  int n  = nt * 16 + (lane & 15);

  v8f c = {};
  const _Float16* arow = A + (mt * 16 + m) * 128;
#pragma unroll
  for (int kt = 0; kt < 4; ++kt) {
    int kbase = kt * 32;
    v16h a, b;
    // A 16x32 f16 layout: halves 0..15 -> K = kbase + hi*16 + h
#pragma unroll
    for (int h = 0; h < 8; ++h) {
      a[h]     = arow[kbase + hi * 16 + h];
      a[h + 8] = arow[kbase + hi * 16 + 8 + h];
    }
    // B 32x16 f16 layout: lane -> N, halves -> K = kbase + hi*16 + h
#pragma unroll
    for (int h = 0; h < 16; ++h)
      b[h] = Bm[(kbase + hi * 16 + h) * 128 + n];
    c = __builtin_amdgcn_wmma_f32_16x16x32_f16(false, a, false, b,
                                               (short)0, c, false, false);
  }
  float bias = b1[n];
#pragma unroll
  for (int r = 0; r < 8; ++r) {
    int row = mt * 16 + r + hi * 8;                // C layout: VGPR r -> M=r / r+8
    H[row * 128 + n] = fmaxf(c[r] + bias, 0.f);
  }
}

// ---------------------------------------------------------------------------
// 5) Layer 2 + sigmoid: one wave per batch row.
// ---------------------------------------------------------------------------
__global__ void mlp2_kernel(const float* __restrict__ H,
                            const float* __restrict__ W2,
                            const float* __restrict__ b2,
                            float* __restrict__ out) {
  int gw   = blockIdx.x * 8 + (threadIdx.x >> 5);  // 0..63
  int lane = threadIdx.x & 31;
  float s = 0.f;
  for (int e = lane; e < 128; e += 32) s += H[gw * 128 + e] * W2[e];
  s = wave_sum(s);
  if (lane == 0) out[gw] = 1.0f / (1.0f + expf(-(s + b2[0])));
}

// ---------------------------------------------------------------------------
// 6) Scalar losses.
// ---------------------------------------------------------------------------
__global__ void tmasks_kernel(const float* __restrict__ mw, float* __restrict__ out) {
  __shared__ float red[8];
  float s = 0.f;
  for (int i = threadIdx.x; i < NPAIR * 128; i += 256) s += fmaxf(mw[i], 0.f);
  s = wave_sum(s);
  if ((threadIdx.x & 31) == 0) red[threadIdx.x >> 5] = s;
  __syncthreads();
  if (threadIdx.x == 0) {
    float t = 0.f;
    for (int i = 0; i < 8; ++i) t += red[i];
    out[0] = t * (1.0f / NPAIR);
  }
}

__global__ void featloss_kernel(const float* __restrict__ f, float* __restrict__ out) {
  __shared__ float red[8];
  const f32x4* p = reinterpret_cast<const f32x4*>(f);
  float s = 0.f;
  for (int i = threadIdx.x; i < (BATCH * NITEM * 128) / 4; i += 256) {
    f32x4 v = p[i];
    s += v.x * v.x + v.y * v.y + v.z * v.z + v.w * v.w;
  }
  s = wave_sum(s);
  if ((threadIdx.x & 31) == 0) red[threadIdx.x >> 5] = s;
  __syncthreads();
  if (threadIdx.x == 0) {
    float t = 0.f;
    for (int i = 0; i < 8; ++i) t += red[i];
    out[0] = sqrtf(t) / sqrtf((float)(BATCH * NITEM));
  }
}

// ---------------------------------------------------------------------------
extern "C" void kernel_launch(void* const* d_in, const int* in_sizes, int n_in,
                              void* d_out, int out_size, void* d_ws, size_t ws_size,
                              hipStream_t stream) {
  const float* features = (const float*)d_in[0];
  const float* rep_l1   = (const float*)d_in[1];
  const float* rep_l2   = (const float*)d_in[2];
  const float* rep_l3   = (const float*)d_in[3];
  const float* masks_w  = (const float*)d_in[4];
  const float* masks_l1 = (const float*)d_in[5];
  const float* masks_l2 = (const float*)d_in[6];
  const float* masks_l3 = (const float*)d_in[7];
  const float* bn_gamma = (const float*)d_in[8];
  const float* bn_beta  = (const float*)d_in[9];
  const float* W1       = (const float*)d_in[10];
  const float* b1       = (const float*)d_in[11];
  const float* W2       = (const float*)d_in[12];
  const float* b2       = (const float*)d_in[13];
  float* out = (float*)d_out;

  // workspace layout (bytes)
  char* ws = (char*)d_ws;
  float* pooled1 = (float*)(ws + 0);                       // 28672 f32
  float* pooled2 = (float*)(ws + 28672u * 4);              // 57344 f32
  float* pooled3 = (float*)(ws + (28672u + 57344u) * 4);   // 114688 f32
  float* rel     = (float*)(ws + 200704u * 4);             // 64*112 f32
  float* H       = (float*)(ws + 207872u * 4);             // 64*128 f32
  _Float16* Astage = (_Float16*)(ws + 216064u * 4);        // 64*128 f16
  _Float16* W1h    = (_Float16*)(ws + 216064u * 4 + 16384u); // 128*128 f16

  // 1) bandwidth-bound pooling (one wave per plane; 8 waves/block)
  pool_kernel<<<28672 / 8, 256, 0, stream>>>(rep_l1, pooled1, 28672, 784, 1.0f / 3136.0f);
  pool_kernel<<<57344 / 8, 256, 0, stream>>>(rep_l2, pooled2, 57344, 196, 1.0f / 784.0f);
  pool_kernel<<<114688 / 8, 256, 0, stream>>>(rep_l3, pooled3, 114688, 49, 1.0f / 196.0f);

  // 2) pair cosines: 64*28 waves per stage = 224 blocks of 256
  paircos_kernel<<<224, 256, 0, stream>>>(features, masks_w, rel, 128, 1, 0);
  paircos_kernel<<<224, 256, 0, stream>>>(pooled1, masks_l1, rel, 64, 0, 28);
  paircos_kernel<<<224, 256, 0, stream>>>(pooled2, masks_l2, rel, 128, 0, 56);
  paircos_kernel<<<224, 256, 0, stream>>>(pooled3, masks_l3, rel, 256, 0, 84);

  // 3) batchnorm + fp16 staging
  bn_stage_kernel<<<1, 256, 0, stream>>>(rel, bn_gamma, bn_beta, W1, Astage, W1h);

  // 4) WMMA MLP layer 1 (32 waves = 4 blocks x 8 waves)
  mlp1_wmma_kernel<<<4, 256, 0, stream>>>(Astage, W1h, b1, H);

  // 5) layer 2 + sigmoid (64 waves)
  mlp2_kernel<<<8, 256, 0, stream>>>(H, W2, b2, out);

  // 6) scalar losses
  tmasks_kernel<<<1, 256, 0, stream>>>(masks_w, out + 64);
  featloss_kernel<<<1, 256, 0, stream>>>(features, out + 65);
}
